// LlamaAttention_29386166239748
// MI455X (gfx1250) — matrix operand order repaired and checked by reading, not compiled, not run on hardware
//
#include <hip/hip_runtime.h>
#include <hip/hip_bf16.h>

// ---------------------------------------------------------------------------
// LLaMA attention layer on gfx1250 (MI455X): bf16 WMMA everywhere.
//   B=2 S=2048 H=32 KV=8 D=128 DM=4096
// Pipeline: cvt f32->bf16 -> [QKV GEMM + RoPE] -> flash attn -> O-proj GEMM
// GEMMs: double-buffered LDS, global->reg prefetch, rotating fragment
// buffers so ds_load latency overlaps the WMMA pipe (partial dscnt waits).
// ---------------------------------------------------------------------------

#define B_    2
#define S_    2048
#define H_    32
#define KV_   8
#define D_    128
#define DM_   4096
#define T_    (B_ * S_)              // 4096
#define NQKV_ ((H_ + 2 * KV_) * D_)  // 6144

typedef __attribute__((ext_vector_type(16))) __bf16   v16bf;
typedef __attribute__((ext_vector_type(8)))  float    v8f;
typedef __attribute__((ext_vector_type(4)))  unsigned v4u;

static __device__ inline int lane_id() { return (int)(threadIdx.x & 31u); }

// A-fragment 16x32 (MxK) bf16 from row-major storage, leading dim ld (elems).
// Layout: lanes 0-15 -> M=lane, K=0..15; lanes 16-31 -> same M,
// K=16..31.  Two 16-byte loads per lane.
static __device__ inline v16bf load_a_frag(const __bf16* base, int ld) {
  const int l = lane_id();
  const int m = l & 15;
  const int g = (l >> 4) * 16;
  const __bf16* p = base + m * ld + g;
  v16bf f;
  ((v4u*)&f)[0] = *(const v4u*)(p);        // K = g .. g+7
  ((v4u*)&f)[1] = *(const v4u*)(p + 8);    // K = g+8 .. g+15
  return f;
}

// B-fragment 32x16 (KxN) bf16 loaded from N-major (transposed) storage
// Bt[n,k], leading dim ld. Layout used here (contiguous k, as A): lanes 0-15 -> N=lane, K=0..15;
// lanes 16-31 -> N=lane-16, K=16..31. 16 contiguous bf16 per lane.
static __device__ inline v16bf load_b_frag(const __bf16* baseT, int ld) {
  const int l = lane_id();
  const int n = l & 15;
  const int g = (l >> 4) * 16;
  const __bf16* p = baseT + n * ld + g;
  v16bf f;
  ((v4u*)&f)[0] = *(const v4u*)(p);
  ((v4u*)&f)[1] = *(const v4u*)(p + 8);
  return f;
}

static __device__ inline v8f wmma_bf16(v16bf a, v16bf b, v8f c) {
  return __builtin_amdgcn_wmma_f32_16x16x32_bf16(false, a, false, b,
                                                 (short)0, c, false, false);
}

// ---------------------------------------------------------------------------
__global__ void f32_to_bf16_kernel(const float* __restrict__ src,
                                   __bf16* __restrict__ dst, int n) {
  int i = blockIdx.x * blockDim.x + threadIdx.x;
  const int stride = gridDim.x * blockDim.x;
  for (; i < n; i += stride) dst[i] = (__bf16)src[i];
}

// ---------------------------------------------------------------------------
// Shared GEMM core: C(128x128) += X(rows row0..row0+127, K=DM) * W^T
// (cols n0..n0+127).  256 threads / 8 waves, each wave owns 16 rows x 128
// cols (8 f32 accumulators).  K staged in 64-wide double-buffered LDS tiles;
// next tile is prefetched global->registers while current tile computes.
// One workgroup barrier per K-step; 16 WMMA per barrier; B-fragments flow
// through a 4-deep rotating register pipeline.
static __device__ inline void gemm_core(const __bf16* __restrict__ X,
                                        const __bf16* __restrict__ W,
                                        __bf16* ldsA, __bf16* ldsB, int row0,
                                        int n0, int wid, v8f acc[8]) {
  const int tid = threadIdx.x;
  const int sr = tid & 127;          // staging row 0..127
  const int sc = (tid >> 7) * 32;    // staging k offset 0 or 32
  const __bf16* gx = X + (size_t)(row0 + sr) * DM_ + sc;
  const __bf16* gw = W + (size_t)(n0 + sr) * DM_ + sc;
  __bf16* la = ldsA + sr * 64 + sc;
  __bf16* lb = ldsB + sr * 64 + sc;

  v4u pa[4], pb[4];
#pragma unroll
  for (int u = 0; u < 4; ++u) {
    pa[u] = *(const v4u*)(gx + u * 8);
    pb[u] = *(const v4u*)(gw + u * 8);
  }

  const int NT = DM_ / 64;  // 64 K-steps
  for (int t = 0; t < NT; ++t) {
    const int buf = (t & 1) * (128 * 64);
#pragma unroll
    for (int u = 0; u < 4; ++u) {
      *(v4u*)(la + buf + u * 8) = pa[u];
      *(v4u*)(lb + buf + u * 8) = pb[u];
    }
    if (t + 1 < NT) {
      const int ko = (t + 1) * 64;
#pragma unroll
      for (int u = 0; u < 4; ++u) {
        pa[u] = *(const v4u*)(gx + ko + u * 8);
        pb[u] = *(const v4u*)(gw + ko + u * 8);
      }
    }
    __syncthreads();
    const __bf16* A = ldsA + buf + wid * 16 * 64;
    const __bf16* Bt = ldsB + buf;
    v16bf a0 = load_a_frag(A, 64);
    v16bf a1 = load_a_frag(A + 32, 64);
    // 4-deep rotating B-fragment pipeline; schedule order:
    // s = 0..15 -> (n8 = s>>1, kk = (s&1)*32), frag slot s&3.
    v16bf bq[4];
    bq[0] = load_b_frag(Bt + 0 * 16 * 64 + 0, 64);
    bq[1] = load_b_frag(Bt + 0 * 16 * 64 + 32, 64);
    bq[2] = load_b_frag(Bt + 1 * 16 * 64 + 0, 64);
    bq[3] = load_b_frag(Bt + 1 * 16 * 64 + 32, 64);
#pragma unroll
    for (int s = 0; s < 16; ++s) {
      const int n8 = s >> 1;
      acc[n8] = wmma_bf16((s & 1) ? a1 : a0, bq[s & 3], acc[n8]);
      const int p = s + 4;
      if (p < 16)
        bq[s & 3] = load_b_frag(Bt + (p >> 1) * 16 * 64 + (p & 1) * 32, 64);
    }
    // No trailing barrier: double-buffered, a wave can be at most one
    // iteration behind the barrier above, which only touches the other buffer.
  }
}

// ---------------------------------------------------------------------------
// GEMM1: qkv = X(T x DM) * W(NQKV x DM)^T, fused RoPE epilogue.
// blockIdx.x = head-column (0..47): 128 N-cols = one head.
__global__ __launch_bounds__(256) void qkv_rope_gemm(
    const __bf16* __restrict__ X, const __bf16* __restrict__ W,
    const float* __restrict__ cosp, const float* __restrict__ sinp,
    __bf16* __restrict__ qb, __bf16* __restrict__ kbuf,
    __bf16* __restrict__ vt) {
  __shared__ __align__(16) __bf16 ldsA[2 * 128 * 64];
  __shared__ __align__(16) __bf16 ldsB[2 * 128 * 64];
  const int wid = threadIdx.x >> 5;
  const int l = lane_id();
  const int row0 = blockIdx.y * 128;
  const int n0 = blockIdx.x * 128;

  v8f acc[8];
#pragma unroll
  for (int t = 0; t < 8; ++t)
#pragma unroll
    for (int r = 0; r < 8; ++r) acc[t][r] = 0.0f;

  gemm_core(X, W, ldsA, ldsB, row0, n0, wid, acc);

  // Epilogue: C(m,n): vgpr r, lane -> m = r + 8*(lane>=16), n = lane&15.
  const int half = l >> 4, nl = l & 15;
  const int nt = blockIdx.x;  // 0..31 q heads, 32..39 k heads, 40..47 v heads
#pragma unroll
  for (int r = 0; r < 8; ++r) {
    const int row = row0 + wid * 16 + r + 8 * half;  // token index
    if (nt < 40) {
      // RoPE: out[d] = x[d]*cos - x[d+64]*sin ; out[d+64] = x[d+64]*cos + x[d]*sin
#pragma unroll
      for (int t = 0; t < 4; ++t) {
        const int d1 = t * 16 + nl, d2 = d1 + 64;
        const float x1 = acc[t][r], x2 = acc[t + 4][r];
        const float o1 = x1 * cosp[row * D_ + d1] - x2 * sinp[row * D_ + d1];
        const float o2 = x2 * cosp[row * D_ + d2] + x1 * sinp[row * D_ + d2];
        if (nt < 32) {
          qb[(row * H_ + nt) * D_ + d1] = (__bf16)o1;
          qb[(row * H_ + nt) * D_ + d2] = (__bf16)o2;
        } else {
          kbuf[(row * KV_ + (nt - 32)) * D_ + d1] = (__bf16)o1;
          kbuf[(row * KV_ + (nt - 32)) * D_ + d2] = (__bf16)o2;
        }
      }
    } else {
      // V: store transposed  vt[((b*KV+kv)*D + d)*S + s]
      const int kvh = nt - 40;
      const int bb = row >> 11, ss = row & (S_ - 1);
#pragma unroll
      for (int t = 0; t < 8; ++t) {
        const int d = t * 16 + nl;
        vt[((bb * KV_ + kvh) * D_ + d) * S_ + ss] = (__bf16)acc[t][r];
      }
    }
  }
}

// ---------------------------------------------------------------------------
// Flash attention. One wave = 16 query rows.  Computes S^T = K*Q^T so each
// q-row's softmax stats live in one lane column; PV done as O^T = V^T * P^T.
// K- and V-fragments flow through 2-deep rotating register pipelines.
__global__ __launch_bounds__(128) void flash_attn(
    const __bf16* __restrict__ qb, const __bf16* __restrict__ kbuf,
    const __bf16* __restrict__ vt, __bf16* __restrict__ ao) {
  const int wid = threadIdx.x >> 5;
  const int l = lane_id();
  const int half = l >> 4, nl = l & 15;
  const int b = blockIdx.x / H_, h = blockIdx.x % H_;
  const int kv = h / (H_ / KV_);
  const int q0 = blockIdx.y * 64 + wid * 16;
  const float scale = 0.08838834764831845f;  // D^-0.5
  const int qrow = q0 + nl;

  // Q as B-fragments: Bt[n=qrow, k=d], ld = H*D
  const __bf16* qbase = qb + ((size_t)(b * S_ + q0) * H_ + h) * D_;
  v16bf qf[4];
#pragma unroll
  for (int c = 0; c < 4; ++c) qf[c] = load_b_frag(qbase + c * 32, H_ * D_);

  v8f ot[8];
#pragma unroll
  for (int t = 0; t < 8; ++t)
#pragma unroll
    for (int r = 0; r < 8; ++r) ot[t][r] = 0.0f;
  float m_i = -3.0e38f, l_i = 0.0f;

  for (int kt0 = 0; kt0 < q0 + 16; kt0 += 32) {
    // ---- scores S^T (two 16-token tiles), contraction over D=128 ----
    const __bf16* kbase = kbuf + ((size_t)(b * S_ + kt0) * KV_ + kv) * D_;
    const __bf16* kbase2 = kbase + 16 * KV_ * D_;
    v8f s0, s1;
#pragma unroll
    for (int r = 0; r < 8; ++r) { s0[r] = 0.0f; s1[r] = 0.0f; }
    // schedule: s = 0..7 -> (tile = s>>2, chunk = s&3), frag slot s&1.
    v16bf kq[2];
    kq[0] = load_a_frag(kbase, KV_ * D_);
    kq[1] = load_a_frag(kbase + 32, KV_ * D_);
#pragma unroll
    for (int s = 0; s < 8; ++s) {
      if (s < 4) s0 = wmma_bf16(kq[s & 1], qf[s & 3], s0);
      else       s1 = wmma_bf16(kq[s & 1], qf[s & 3], s1);
      const int p = s + 2;
      if (p < 8)
        kq[s & 1] = load_a_frag(((p >> 2) ? kbase2 : kbase) + (p & 3) * 32,
                                KV_ * D_);
    }
    // ---- scale + causal mask + online softmax ----
    float p0[8], p1[8];
    float mloc = -3.0e38f;
#pragma unroll
    for (int r = 0; r < 8; ++r) {
      const int t0 = kt0 + r + 8 * half;
      const float v0 = (t0 <= qrow) ? s0[r] * scale : -3.0e38f;
      const float v1 = (t0 + 16 <= qrow) ? s1[r] * scale : -3.0e38f;
      p0[r] = v0; p1[r] = v1;
      mloc = fmaxf(mloc, fmaxf(v0, v1));
    }
    mloc = fmaxf(mloc, __shfl_xor(mloc, 16));
    const float mnew = fmaxf(m_i, mloc);
    const float corr = __expf(m_i - mnew);
    float rs = 0.0f;
#pragma unroll
    for (int r = 0; r < 8; ++r) {
      p0[r] = __expf(p0[r] - mnew);
      p1[r] = __expf(p1[r] - mnew);
      rs += p0[r] + p1[r];
    }
    rs += __shfl_xor(rs, 16);
    l_i = l_i * corr + rs;
    m_i = mnew;
#pragma unroll
    for (int t = 0; t < 8; ++t)
#pragma unroll
      for (int r = 0; r < 8; ++r) ot[t][r] = ot[t][r] * corr;

    // ---- build P^T B-fragment (32 tokens x 16 q) from the S^T tiles ----
    // B lanes 0-15 need tokens 0..15 (tile s0), lanes 16-31 tokens 16..31 (s1).
    float o0[8], o1[8];
#pragma unroll
    for (int r = 0; r < 8; ++r) {
      o0[r] = __shfl_xor(p0[r], 16);
      o1[r] = __shfl_xor(p1[r], 16);
    }
    v16bf bp;
#pragma unroll
    for (int e = 0; e < 16; ++e) {
      float v;
      if (half == 0) v = (e < 8) ? p0[e] : o0[e - 8];
      else           v = (e < 8) ? o1[e] : p1[e - 8];
      bp[e] = (__bf16)v;
    }
    // ---- O^T += V^T * P^T, eight 16-d tiles, 2-deep V-frag rotation ----
    const __bf16* vtb = vt + ((size_t)(b * KV_ + kv) * D_) * S_ + kt0;
    v16bf vq[2];
    vq[0] = load_a_frag(vtb, S_);
    vq[1] = load_a_frag(vtb + (size_t)16 * S_, S_);
#pragma unroll
    for (int dt = 0; dt < 8; ++dt) {
      ot[dt] = wmma_bf16(vq[dt & 1], bp, ot[dt]);
      if (dt + 2 < 8)
        vq[dt & 1] = load_a_frag(vtb + (size_t)((dt + 2) * 16) * S_, S_);
    }
  }

  const float linv = 1.0f / l_i;  // diagonal always unmasked -> l_i > 0
#pragma unroll
  for (int dt = 0; dt < 8; ++dt)
#pragma unroll
    for (int r = 0; r < 8; ++r) {
      const int d = dt * 16 + r + 8 * half;
      ao[((size_t)(b * S_ + q0 + nl) * H_ + h) * D_ + d] =
          (__bf16)(ot[dt][r] * linv);
    }
}

// ---------------------------------------------------------------------------
// GEMM2: out = AO(T x DM) * Wo(DM x DM)^T, f32 output.
__global__ __launch_bounds__(256) void out_proj_gemm(
    const __bf16* __restrict__ X, const __bf16* __restrict__ W,
    float* __restrict__ out) {
  __shared__ __align__(16) __bf16 ldsA[2 * 128 * 64];
  __shared__ __align__(16) __bf16 ldsB[2 * 128 * 64];
  const int wid = threadIdx.x >> 5;
  const int l = lane_id();
  const int row0 = blockIdx.y * 128;
  const int n0 = blockIdx.x * 128;

  v8f acc[8];
#pragma unroll
  for (int t = 0; t < 8; ++t)
#pragma unroll
    for (int r = 0; r < 8; ++r) acc[t][r] = 0.0f;

  gemm_core(X, W, ldsA, ldsB, row0, n0, wid, acc);

  const int half = l >> 4, nl = l & 15;
#pragma unroll
  for (int r = 0; r < 8; ++r) {
    const int row = row0 + wid * 16 + r + 8 * half;
#pragma unroll
    for (int t = 0; t < 8; ++t)
      out[(size_t)row * DM_ + n0 + t * 16 + nl] = acc[t][r];
  }
}

// ---------------------------------------------------------------------------
extern "C" void kernel_launch(void* const* d_in, const int* in_sizes, int n_in,
                              void* d_out, int out_size, void* d_ws,
                              size_t ws_size, hipStream_t stream) {
  const float* hs   = (const float*)d_in[0];
  const float* cosp = (const float*)d_in[1];
  const float* sinp = (const float*)d_in[2];
  const float* wqkv = (const float*)d_in[3];
  const float* wo   = (const float*)d_in[4];
  float* out = (float*)d_out;

  char* ws = (char*)d_ws;
  size_t off = 0;
  __bf16* hb   = (__bf16*)(ws + off); off += (size_t)T_ * DM_ * 2;       // 32 MB
  __bf16* wqb  = (__bf16*)(ws + off); off += (size_t)NQKV_ * DM_ * 2;    // 48 MB
  __bf16* wob  = (__bf16*)(ws + off); off += (size_t)DM_ * DM_ * 2;      // 32 MB
  __bf16* qbuf = (__bf16*)(ws + off); off += (size_t)T_ * H_ * D_ * 2;   // 32 MB
  __bf16* kbuf = (__bf16*)(ws + off); off += (size_t)T_ * KV_ * D_ * 2;  //  8 MB
  __bf16* vtb  = (__bf16*)(ws + off); off += (size_t)T_ * KV_ * D_ * 2;  //  8 MB
  __bf16* aob  = (__bf16*)(ws + off);                                    // 32 MB

  f32_to_bf16_kernel<<<1024, 256, 0, stream>>>(hs, hb, T_ * DM_);
  f32_to_bf16_kernel<<<1024, 256, 0, stream>>>(wqkv, wqb, NQKV_ * DM_);
  f32_to_bf16_kernel<<<1024, 256, 0, stream>>>(wo, wob, DM_ * DM_);

  qkv_rope_gemm<<<dim3(NQKV_ / 128, T_ / 128), 256, 0, stream>>>(
      hb, wqb, cosp, sinp, qbuf, kbuf, vtb);

  flash_attn<<<dim3(B_ * H_, S_ / 64), 128, 0, stream>>>(qbuf, kbuf, vtb, aob);

  out_proj_gemm<<<dim3(DM_ / 128, T_ / 128), 256, 0, stream>>>(aob, wob, out);
}
